// MNDRE_40596030882271
// MI455X (gfx1250) — matrix operand order repaired and checked by reading, not compiled, not run on hardware
//
#include <hip/hip_runtime.h>
#include <math.h>

// ---------------- problem constants ----------------
#define D_      3
#define H_      64
#define WIN_    8
#define OUT_    32
#define B_      128
#define T_      129
#define NW_     16
#define NSTEP_  128
#define NB_     16           // batch elements per workgroup
#define NWG_    (B_/NB_)     // 8 workgroups
#define NTHR_   256          // 8 wave32s

typedef __attribute__((ext_vector_type(16))) __bf16 v16bf;
typedef __attribute__((ext_vector_type(8)))  float  v8f;

// ---------------- dynamic-LDS layout (bytes) ----------------
// f32 section
constexpr size_t OFF_C0  = 0;                   // 128
constexpr size_t OFF_C1  = OFF_C0  + 128*4;
constexpr size_t OFF_C2  = OFF_C1  + 128*4;     // 192
constexpr size_t OFF_RB  = OFF_C2  + 192*4;     // 32
constexpr size_t OFF_Y   = OFF_RB  + 32*4;      // 64*16   state y (f32 master)
constexpr size_t OFF_SG0 = OFF_Y   + 1024*4;    // 128*16  sigmoid(u0)
constexpr size_t OFF_SG1 = OFF_SG0 + 2048*4;    // 128*16  sigmoid(u1)
constexpr size_t OFF_T2  = OFF_SG1 + 2048*4;    // 192*16  tanh(u2) = m
constexpr size_t OFF_J   = OFF_T2  + 3072*4;    // 3*192*16 JVP outputs
constexpr size_t OFF_K1  = OFF_J   + 9216*4;    // 64*16
constexpr size_t OFF_KV  = OFF_K1  + 1024*4;    // 64*16
constexpr size_t OFF_S   = OFF_KV  + 1024*4;    // 16*6
constexpr size_t OFF_DT  = OFF_S   + 96*4;      // 16
// bf16 weight section (16B aligned, row-major for A-fragments)
constexpr size_t OFF_V0  = OFF_DT  + 16*4;      // 128*64
constexpr size_t OFF_V1  = OFF_V0  + 8192*2;    // 128*128
constexpr size_t OFF_V2  = OFF_V1  + 16384*2;   // 192*128
constexpr size_t OFF_RW  = OFF_V2  + 24576*2;   // 32*64  readout R (bf16)
// K-pair-packed bf16 activation buffers (uint32 words: rows (2p,2p+1) x col)
constexpr size_t OFF_YB  = OFF_RW  + 2048*2;    // 512  words (64x16)
constexpr size_t OFF_A0  = OFF_YB  + 512*4;     // 1024 words (128x16)
constexpr size_t OFF_A1  = OFF_A0  + 1024*4;    // 1024 words
constexpr size_t OFF_M   = OFF_A1  + 1024*4;    // 1536 words (192x16)
constexpr size_t OFF_DA0 = OFF_M   + 1536*4;    // 3072 words (128x48)
constexpr size_t OFF_DA1 = OFF_DA0 + 3072*4;    // 3072 words
constexpr size_t SH_BYTES= OFF_DA1 + 3072*4;    // ~218 KB < 320 KB/WGP

struct Shm {
  float    *C0,*C1,*C2,*Rb,*Y,*SG0,*SG1,*T2,*J,*K1,*Kv,*S,*Dt;
  __bf16   *V0,*V1,*V2,*Rw;
  uint32_t *Yb,*A0,*A1,*M,*DA0,*DA1;   // packed-pair bf16 words
};

// ---------------- fast math on the native TRANS pipe ----------------
__device__ __forceinline__ float tanh_f(float x) {
#if __has_builtin(__builtin_amdgcn_tanhf)
  return __builtin_amdgcn_tanhf(x);
#else
  float r;
  asm volatile("v_tanh_f32 %0, %1\n\tv_nop\n\tv_nop" : "=v"(r) : "v"(x));
  return r;
#endif
}
__device__ __forceinline__ float sp_f(float x) {       // softplus
  float r = __logf(1.f + __expf(x));
  return (x > 20.f) ? x : r;                            // branchless cndmask
}
__device__ __forceinline__ float sig_f(float x) {      // sigmoid
  return __builtin_amdgcn_rcpf(1.f + __expf(-x));
}

__device__ __forceinline__ uint32_t pk2(float lo, float hi) {
  union { __bf16 h[2]; uint32_t w; } u;
  u.h[0] = (__bf16)lo; u.h[1] = (__bf16)hi;
  return u.w;
}

// A-fragment (16x32 bf16, ISA 7.12.2): two 16B chunks per lane -> 2x ds_load_b128
__device__ __forceinline__ v16bf load_A_frag(const __bf16* A, int ldk, int mt, int ko, int lane) {
  const int r    = mt*16 + (lane & 15);
  const int half = lane >> 4;
  const uint4* p = reinterpret_cast<const uint4*>(A + (size_t)r*ldk + ko + half*8);
  union { uint4 q[2]; v16bf v; } u;
  u.q[0] = p[0];        // K = ko + half*8 .. +7
  u.q[1] = p[2];        // K = ko + 16 + half*8 .. +7   (+16 elements = +32B)
  return u.v;
}

// B-fragment (32x16 bf16) from K-pair-packed buffer: 8x ds_load_b32
__device__ __forceinline__ v16bf load_B_frag(const uint32_t* Bp, int ldn, int ko, int n0, int lane) {
  const int c    = n0 + (lane & 15);
  const int half = lane >> 4;
  const uint32_t* p = Bp + (size_t)((ko >> 1) + half*4)*ldn + c;
  union { uint32_t w[8]; v16bf v; } u;
#pragma unroll
  for (int j = 0; j < 8; ++j) u.w[j] = p[(size_t)(j < 4 ? j : j + 4)*ldn];
  return u.v;
}

__device__ __forceinline__ v8f wmma_bf16(v16bf a, v16bf b, v8f c) {
  return __builtin_amdgcn_wmma_f32_16x16x32_bf16(false, a, false, b, (short)0, c, false, false);
}

// ---------------- one F(y) evaluation, 16 batch columns ----------------
__device__ __forceinline__ void evalF(const Shm& sh, int tid) {
  const int wave = tid >> 5, lane = tid & 31;
  const int half = lane >> 4, col = lane & 15;

  // fwd L0: u0 = V0(128x64) @ y ; cache sigmoid(u0), softplus(u0)->A0
  for (int mt = wave; mt < 8; mt += 8) {
    v8f acc = {};
#pragma unroll
    for (int ko = 0; ko < 64; ko += 32)
      acc = wmma_bf16(load_A_frag(sh.V0, 64, mt, ko, lane), load_B_frag(sh.Yb, 16, ko, 0, lane), acc);
    const int r0 = mt*16 + 8*half;
#pragma unroll
    for (int gp = 0; gp < 4; ++gp) {
      float ua = acc[2*gp]   + sh.C0[r0 + 2*gp];
      float ub = acc[2*gp+1] + sh.C0[r0 + 2*gp+1];
      sh.SG0[(r0 + 2*gp)*16 + col]   = sig_f(ua);
      sh.SG0[(r0 + 2*gp+1)*16 + col] = sig_f(ub);
      sh.A0[((r0 >> 1) + gp)*16 + col] = pk2(sp_f(ua), sp_f(ub));
    }
  }
  __syncthreads();

  // fwd L1
  for (int mt = wave; mt < 8; mt += 8) {
    v8f acc = {};
#pragma unroll
    for (int ko = 0; ko < 128; ko += 32)
      acc = wmma_bf16(load_A_frag(sh.V1, 128, mt, ko, lane), load_B_frag(sh.A0, 16, ko, 0, lane), acc);
    const int r0 = mt*16 + 8*half;
#pragma unroll
    for (int gp = 0; gp < 4; ++gp) {
      float ua = acc[2*gp]   + sh.C1[r0 + 2*gp];
      float ub = acc[2*gp+1] + sh.C1[r0 + 2*gp+1];
      sh.SG1[(r0 + 2*gp)*16 + col]   = sig_f(ua);
      sh.SG1[(r0 + 2*gp+1)*16 + col] = sig_f(ub);
      sh.A1[((r0 >> 1) + gp)*16 + col] = pk2(sp_f(ua), sp_f(ub));
    }
  }
  __syncthreads();

  // fwd L2: T2 = tanh(V2 @ A1 + c2), M = bf16(T2) (tangent source)
  for (int mt = wave; mt < 12; mt += 8) {
    v8f acc = {};
#pragma unroll
    for (int ko = 0; ko < 128; ko += 32)
      acc = wmma_bf16(load_A_frag(sh.V2, 128, mt, ko, lane), load_B_frag(sh.A1, 16, ko, 0, lane), acc);
    const int r0 = mt*16 + 8*half;
#pragma unroll
    for (int gp = 0; gp < 4; ++gp) {
      float ta = tanh_f(acc[2*gp]   + sh.C2[r0 + 2*gp]);
      float tb = tanh_f(acc[2*gp+1] + sh.C2[r0 + 2*gp+1]);
      sh.T2[(r0 + 2*gp)*16 + col]   = ta;
      sh.T2[(r0 + 2*gp+1)*16 + col] = tb;
      sh.M[((r0 >> 1) + gp)*16 + col] = pk2(ta, tb);
    }
  }
  __syncthreads();

  // JVP L0: DA0 = sigmoid(u0) * (V0 @ m[nt]) ; tangent nt = N-tile
  for (int task = wave; task < 24; task += 8) {
    const int mt = task & 7, nt = task >> 3;
    v8f acc = {};
#pragma unroll
    for (int ko = 0; ko < 64; ko += 32)
      acc = wmma_bf16(load_A_frag(sh.V0, 64, mt, ko, lane),
                      load_B_frag(sh.M + nt*512, 16, ko, 0, lane), acc);
    const int r0 = mt*16 + 8*half, nc = nt*16 + col;
#pragma unroll
    for (int gp = 0; gp < 4; ++gp) {
      float da = sh.SG0[(r0 + 2*gp)*16 + col]   * acc[2*gp];
      float db = sh.SG0[(r0 + 2*gp+1)*16 + col] * acc[2*gp+1];
      sh.DA0[((r0 >> 1) + gp)*48 + nc] = pk2(da, db);
    }
  }
  __syncthreads();

  // JVP L1
  for (int task = wave; task < 24; task += 8) {
    const int mt = task & 7, nt = task >> 3;
    v8f acc = {};
#pragma unroll
    for (int ko = 0; ko < 128; ko += 32)
      acc = wmma_bf16(load_A_frag(sh.V1, 128, mt, ko, lane),
                      load_B_frag(sh.DA0, 48, ko, nt*16, lane), acc);
    const int r0 = mt*16 + 8*half, nc = nt*16 + col;
#pragma unroll
    for (int gp = 0; gp < 4; ++gp) {
      float da = sh.SG1[(r0 + 2*gp)*16 + col]   * acc[2*gp];
      float db = sh.SG1[(r0 + 2*gp+1)*16 + col] * acc[2*gp+1];
      sh.DA1[((r0 >> 1) + gp)*48 + nc] = pk2(da, db);
    }
  }
  __syncthreads();

  // JVP L2: J[nt] = (1 - T2^2) * (V2 @ DA1)
  for (int task = wave; task < 36; task += 8) {
    const int mt = task % 12, nt = task / 12;
    v8f acc = {};
#pragma unroll
    for (int ko = 0; ko < 128; ko += 32)
      acc = wmma_bf16(load_A_frag(sh.V2, 128, mt, ko, lane),
                      load_B_frag(sh.DA1, 48, ko, nt*16, lane), acc);
    const int r0 = mt*16 + 8*half;
#pragma unroll
    for (int g = 0; g < 8; ++g) {
      float t = sh.T2[(r0 + g)*16 + col];
      sh.J[((size_t)nt*192 + r0 + g)*16 + col] = (1.f - t*t) * acc[g];
    }
  }
  __syncthreads();

  // combine: Kv[h][b] = F(y) @ s   (pairs (0,1),(0,2),(1,2))
  for (int i = tid; i < 64*16; i += NTHR_) {
    const int h = i >> 4, b = i & 15;
    const float* S = sh.S + b*6;
    float m0 = sh.T2[(0*64 + h)*16 + b];
    float m1 = sh.T2[(1*64 + h)*16 + b];
    float m2 = sh.T2[(2*64 + h)*16 + b];
    auto Jat = [&](int tg, int a_) { return sh.J[((size_t)tg*192 + a_*64 + h)*16 + b]; };
    sh.Kv[i] = S[0]*m0 + S[1]*m1 + S[2]*m2
             + S[3]*(Jat(0,1) - Jat(1,0))
             + S[4]*(Jat(0,2) - Jat(2,0))
             + S[5]*(Jat(1,2) - Jat(2,1));
  }
  __syncthreads();
}

// readout row: out[b, trow, :] = tanh(R(32x64) @ y + rb) as a WMMA GEMM (waves 0-1)
__device__ __forceinline__ void emit_row(const Shm& sh, float* __restrict__ out,
                                         int b0, int trow, int tid) {
  const int wave = tid >> 5, lane = tid & 31;
  const int half = lane >> 4, colb = lane & 15;
  for (int mt = wave; mt < 2; mt += 8) {
    v8f acc = {};
#pragma unroll
    for (int ko = 0; ko < 64; ko += 32)
      acc = wmma_bf16(load_A_frag(sh.Rw, 64, mt, ko, lane), load_B_frag(sh.Yb, 16, ko, 0, lane), acc);
    const int o0 = mt*16 + 8*half;
    float t[8];
#pragma unroll
    for (int g = 0; g < 8; ++g) t[g] = tanh_f(acc[g] + sh.Rb[o0 + g]);
    float* dst = out + (((size_t)(b0 + colb))*T_ + trow)*OUT_ + o0;
    *(float4*)(dst)     = make_float4(t[0], t[1], t[2], t[3]);
    *(float4*)(dst + 4) = make_float4(t[4], t[5], t[6], t[7]);
  }
}

// ---------------- kernel 1: log-signatures -> slopes ----------------
__global__ void logsig_kernel(const float* __restrict__ ts, const float* __restrict__ x,
                              float* __restrict__ slws) {
  const int idx = blockIdx.x*blockDim.x + threadIdx.x;
  if (idx >= B_*NW_) return;
  const int b = idx / NW_, w = idx % NW_;
  const float* xb = x + ((size_t)b*T_ + w*WIN_)*D_;
  float w0[3] = { xb[0], xb[1], xb[2] };
  float prev[3] = { w0[0], w0[1], w0[2] };
  float a[3][3] = {};
  for (int k = 0; k < WIN_; ++k) {
    const float* cur = xb + (size_t)(k+1)*D_;
    float rel[3], dv[3];
    for (int d = 0; d < 3; ++d) { rel[d] = prev[d] - w0[d]; dv[d] = cur[d] - prev[d]; }
    for (int i2 = 0; i2 < 3; ++i2)
      for (int j2 = 0; j2 < 3; ++j2) a[i2][j2] += rel[i2]*dv[j2];
    prev[0] = cur[0]; prev[1] = cur[1]; prev[2] = cur[2];
  }
  const float inv = 1.f / (ts[(size_t)b*T_ + (w+1)*WIN_] - ts[(size_t)b*T_ + w*WIN_]);
  float* o = slws + ((size_t)b*NW_ + w)*6;
  o[0] = (prev[0]-w0[0])*inv; o[1] = (prev[1]-w0[1])*inv; o[2] = (prev[2]-w0[2])*inv;
  o[3] = 0.5f*(a[0][1]-a[1][0])*inv;
  o[4] = 0.5f*(a[0][2]-a[2][0])*inv;
  o[5] = 0.5f*(a[1][2]-a[2][1])*inv;
}

// ---------------- kernel 2: init MLP h0 ----------------
__global__ void init_kernel(const float* __restrict__ x,
                            const float* __restrict__ W0, const float* __restrict__ b0,
                            const float* __restrict__ W1, const float* __restrict__ b1,
                            const float* __restrict__ W2, const float* __restrict__ b2,
                            float* __restrict__ h0ws) {
  __shared__ float a0[128], a1[128];
  const int b = blockIdx.x, r = threadIdx.x;
  const float x0 = x[(size_t)b*T_*D_ + 0];
  const float x1 = x[(size_t)b*T_*D_ + 1];
  const float x2 = x[(size_t)b*T_*D_ + 2];
  a0[r] = sp_f(b0[r] + W0[r*3]*x0 + W0[r*3+1]*x1 + W0[r*3+2]*x2);
  __syncthreads();
  float u1 = b1[r];
  for (int k = 0; k < 128; ++k) u1 += W1[r*128 + k]*a0[k];
  a1[r] = sp_f(u1);
  __syncthreads();
  if (r < H_) {
    float h = b2[r];
    for (int k = 0; k < 128; ++k) h += W2[r*128 + k]*a1[k];
    h0ws[(size_t)b*H_ + r] = h;
  }
}

// ---------------- kernel 3: RK2 CDE solve ----------------
__global__ __launch_bounds__(NTHR_, 1)
void solve_kernel(const float* __restrict__ ts,
                  const float* __restrict__ V0g, const float* __restrict__ c0g,
                  const float* __restrict__ V1g, const float* __restrict__ c1g,
                  const float* __restrict__ V2g, const float* __restrict__ c2g,
                  const float* __restrict__ Rg,  const float* __restrict__ rbg,
                  const float* __restrict__ h0ws, const float* __restrict__ slws,
                  float* __restrict__ out) {
  extern __shared__ char smem[];
  Shm sh;
  sh.C0=(float*)(smem+OFF_C0);  sh.C1=(float*)(smem+OFF_C1);  sh.C2=(float*)(smem+OFF_C2);
  sh.Rb=(float*)(smem+OFF_RB);  sh.Y=(float*)(smem+OFF_Y);
  sh.SG0=(float*)(smem+OFF_SG0);sh.SG1=(float*)(smem+OFF_SG1);sh.T2=(float*)(smem+OFF_T2);
  sh.J=(float*)(smem+OFF_J);    sh.K1=(float*)(smem+OFF_K1);  sh.Kv=(float*)(smem+OFF_KV);
  sh.S=(float*)(smem+OFF_S);    sh.Dt=(float*)(smem+OFF_DT);
  sh.V0=(__bf16*)(smem+OFF_V0); sh.V1=(__bf16*)(smem+OFF_V1); sh.V2=(__bf16*)(smem+OFF_V2);
  sh.Rw=(__bf16*)(smem+OFF_RW);
  sh.Yb=(uint32_t*)(smem+OFF_YB); sh.A0=(uint32_t*)(smem+OFF_A0); sh.A1=(uint32_t*)(smem+OFF_A1);
  sh.M=(uint32_t*)(smem+OFF_M);   sh.DA0=(uint32_t*)(smem+OFF_DA0); sh.DA1=(uint32_t*)(smem+OFF_DA1);

  const int tid = threadIdx.x;
  const int b0  = blockIdx.x * NB_;

  for (int i = tid; i < 128*64;  i += NTHR_) sh.V0[i] = (__bf16)V0g[i];
  for (int i = tid; i < 128*128; i += NTHR_) sh.V1[i] = (__bf16)V1g[i];
  for (int i = tid; i < 192*128; i += NTHR_) sh.V2[i] = (__bf16)V2g[i];
  for (int i = tid; i < 32*64;   i += NTHR_) sh.Rw[i] = (__bf16)Rg[i];
  if (tid < 128) { sh.C0[tid] = c0g[tid]; sh.C1[tid] = c1g[tid]; }
  if (tid < 192) sh.C2[tid] = c2g[tid];
  if (tid < OUT_) sh.Rb[tid] = rbg[tid];
  // h0 (column-per-batch, packed pairs for the B-matrix copy)
  for (int i = tid; i < 512; i += NTHR_) {
    const int hp = i >> 4, b = i & 15;
    float va = h0ws[(size_t)(b0 + b)*H_ + 2*hp];
    float vb = h0ws[(size_t)(b0 + b)*H_ + 2*hp + 1];
    sh.Y[(2*hp)*16 + b] = va; sh.Y[(2*hp+1)*16 + b] = vb;
    sh.Yb[hp*16 + b] = pk2(va, vb);
  }
  __syncthreads();

  emit_row(sh, out, b0, 0, tid);            // hidden row 0 = h0

  for (int step = 0; step < NSTEP_; ++step) {
    if (tid < NB_*6) {
      const int b = tid / 6, j = tid - b*6;
      sh.S[b*6 + j] = slws[((size_t)(b0 + b)*NW_ + (step >> 3))*6 + j];
    }
    if (tid < NB_)
      sh.Dt[tid] = ts[(size_t)(b0 + tid)*T_ + step + 1] - ts[(size_t)(b0 + tid)*T_ + step];
    __syncthreads();

    evalF(sh, tid);                          // k1 -> Kv
    for (int i = tid; i < 512; i += NTHR_) {
      const int hp = i >> 4, b = i & 15;
      const int i0 = (2*hp)*16 + b, i1 = i0 + 16;
      float k1a = sh.Kv[i0], k1b = sh.Kv[i1];
      sh.K1[i0] = k1a; sh.K1[i1] = k1b;
      sh.Yb[hp*16 + b] = pk2(sh.Y[i0] + sh.Dt[b]*k1a, sh.Y[i1] + sh.Dt[b]*k1b);
    }
    __syncthreads();

    evalF(sh, tid);                          // k2 -> Kv
    for (int i = tid; i < 512; i += NTHR_) {
      const int hp = i >> 4, b = i & 15;
      const int i0 = (2*hp)*16 + b, i1 = i0 + 16;
      float ya = sh.Y[i0] + 0.5f*sh.Dt[b]*(sh.K1[i0] + sh.Kv[i0]);
      float yb = sh.Y[i1] + 0.5f*sh.Dt[b]*(sh.K1[i1] + sh.Kv[i1]);
      sh.Y[i0] = ya; sh.Y[i1] = yb;
      sh.Yb[hp*16 + b] = pk2(ya, yb);
    }
    __syncthreads();

    emit_row(sh, out, b0, step + 1, tid);
    __syncthreads();
  }
}

// ---------------- host entry ----------------
extern "C" void kernel_launch(void* const* d_in, const int* in_sizes, int n_in,
                              void* d_out, int out_size, void* d_ws, size_t ws_size,
                              hipStream_t stream) {
  const float* ts = (const float*)d_in[0];
  const float* x  = (const float*)d_in[1];
  const float* W0 = (const float*)d_in[2];
  const float* b0 = (const float*)d_in[3];
  const float* W1 = (const float*)d_in[4];
  const float* b1 = (const float*)d_in[5];
  const float* W2 = (const float*)d_in[6];
  const float* b2 = (const float*)d_in[7];
  const float* V0 = (const float*)d_in[8];
  const float* c0 = (const float*)d_in[9];
  const float* V1 = (const float*)d_in[10];
  const float* c1 = (const float*)d_in[11];
  const float* V2 = (const float*)d_in[12];
  const float* c2 = (const float*)d_in[13];
  const float* R  = (const float*)d_in[14];
  const float* rb = (const float*)d_in[15];
  float* out = (float*)d_out;

  float* h0ws = (float*)d_ws;             // B*H floats
  float* slws = h0ws + (size_t)B_*H_;     // B*NW*6 floats

  hipLaunchKernelGGL(logsig_kernel, dim3((B_*NW_ + 255)/256), dim3(256), 0, stream, ts, x, slws);
  hipLaunchKernelGGL(init_kernel,   dim3(B_), dim3(128), 0, stream, x, W0, b0, W1, b1, W2, b2, h0ws);

  (void)hipFuncSetAttribute((const void*)solve_kernel,
                            hipFuncAttributeMaxDynamicSharedMemorySize, (int)SH_BYTES);
  hipLaunchKernelGGL(solve_kernel, dim3(NWG_), dim3(NTHR_), SH_BYTES, stream,
                     ts, V0, c0, V1, c1, V2, c2, R, rb, h0ws, slws, out);
  (void)in_sizes; (void)n_in; (void)out_size; (void)ws_size;
}
